// UpdateNodeInFrame_85744727097801
// MI455X (gfx1250) — matrix operand order repaired and checked by reading, not compiled, not run on hardware
//
#include <hip/hip_runtime.h>
#include <hip/hip_bf16.h>
#include <math.h>

#define NODES 20000
#define EDGES 100000
#define NDIM  240
#define LATD  128
#define OHD   128

typedef __attribute__((ext_vector_type(16))) _Float16 v16h;
typedef __attribute__((ext_vector_type(8)))  float    v8f;

// ---- f16 weight pool offsets (in halfs), all layouts [N][KPAD] (out-major) ----
constexpr int OFF_W0   = 0;        // 160 x 480
constexpr int OFF_W1R  = 76800;    // 48 x 160
constexpr int OFF_W1I  = 84480;    // 48 x 160
constexpr int OFF_W1IN = 92160;    // 48 x 160 (negated w1i)
constexpr int OFF_W2R  = 99840;    // 16 x 64
constexpr int OFF_W2I  = 100864;   // 16 x 64
constexpr int OFF_W2IN = 101888;   // 16 x 64 (negated w2i)
constexpr int OFF_PW0  = 102912;   // 64 x 64
constexpr int OFF_PW1  = 107008;   // 32 x 32
constexpr int OFF_PW2  = 108032;   // 16 x 32
constexpr int OFF_M1W1 = 108544;   // 128 x 192
constexpr int OFF_M1W2 = 133120;   // 128 x 128
constexpr int OFF_M1W3 = 149504;   // 128 x 128
constexpr int OFF_M2W1 = 165888;   // 128 x 256
constexpr int OFF_M2W2 = 198656;   // 128 x 128
constexpr int OFF_M2W3 = 215040;   // 128 x 128
constexpr int OFF_ENV  = 231424;   // 112 x 128
constexpr int WPOOL_HALFS = 245760;

// ================= WMMA fragment helpers (ISA 7.12.2 layouts) =================
__device__ __forceinline__ v16h load_a_frag(const _Float16* A, int Astride, int k0, int lane) {
  int m = lane & 15, h = lane >> 4;
  const _Float16* row = A + (size_t)m * Astride + k0 + (h << 3);
  v16h a;
#pragma unroll
  for (int v = 0; v < 8; ++v) {
    int k = ((v & 4) << 2) + ((v & 3) << 1);   // v<4: 0..7 ; v>=4: 16..23 (rel to half)
    a[2 * v]     = row[k];
    a[2 * v + 1] = row[k + 1];
  }
  return a;
}

__device__ __forceinline__ v16h load_b_frag(const _Float16* W, int KPAD, int n0, int k0, int lane) {
  int n = (lane & 15) + n0, h = lane >> 4;
  const _Float16* q = W + (size_t)n * KPAD + k0 + (h << 3);   // 8 halfs at 8h, 8 halfs at 16+8h
  v16h b;
#pragma unroll
  for (int i = 0; i < 16; ++i) b[i] = q[i < 8 ? i : i + 8];
  return b;
}

__device__ __forceinline__ void store_c(float* R, int Rstride, int n0, const v8f& c, int lane) {
  int n = (lane & 15) + n0, h = lane >> 4;
#pragma unroll
  for (int r = 0; r < 8; ++r) R[(size_t)(r + 8 * h) * Rstride + n] = c[r];
}

// D[16xN] = A[16xKPAD] * W^T  (W stored [N][KPAD]); optional bias per n
__device__ __forceinline__ void gemm(const _Float16* A, int Astride,
                                     const _Float16* W, int KPAD,
                                     float* R, int Rstride, int N,
                                     const float* bias, int lane) {
  for (int n0 = 0; n0 < N; n0 += 16) {
    v8f c = {};
    for (int k0 = 0; k0 < KPAD; k0 += 32) {
      v16h a = load_a_frag(A, Astride, k0, lane);
      v16h b = load_b_frag(W, KPAD, n0, k0, lane);
      c = __builtin_amdgcn_wmma_f32_16x16x32_f16(false, a, false, b, (short)0, c, false, false);
    }
    if (bias) {
      float bv = bias[(lane & 15) + n0];
#pragma unroll
      for (int r = 0; r < 8; ++r) c[r] += bv;
    }
    store_c(R, Rstride, n0, c, lane);
  }
}

// D = A1*W1^T + A2*W2^T  (complex SO(2) pairs, W2 pre-negated when subtracting)
__device__ __forceinline__ void gemm2(const _Float16* A1, const _Float16* A2, int Astride,
                                      const _Float16* W1, const _Float16* W2, int KPAD,
                                      float* R, int Rstride, int N, int lane) {
  for (int n0 = 0; n0 < N; n0 += 16) {
    v8f c = {};
    for (int k0 = 0; k0 < KPAD; k0 += 32) {
      v16h a1 = load_a_frag(A1, Astride, k0, lane);
      v16h b1 = load_b_frag(W1, KPAD, n0, k0, lane);
      c = __builtin_amdgcn_wmma_f32_16x16x32_f16(false, a1, false, b1, (short)0, c, false, false);
      v16h a2 = load_a_frag(A2, Astride, k0, lane);
      v16h b2 = load_b_frag(W2, KPAD, n0, k0, lane);
      c = __builtin_amdgcn_wmma_f32_16x16x32_f16(false, a2, false, b2, (short)0, c, false, false);
    }
    store_c(R, Rstride, n0, c, lane);
  }
}

__device__ __forceinline__ float siluf(float x) { return x / (1.f + __expf(-x)); }
__device__ __forceinline__ float sigf(float x)  { return 1.f / (1.f + __expf(-x)); }

// ================= K0: convert/transpose/pad weights to f16 [N][KPAD] =========
__global__ void prep_w(const float* __restrict__ src, _Float16* __restrict__ dst,
                       int N, int K, int KPAD, int transpose, float scale) {
  int total = N * KPAD;
  for (int idx = blockIdx.x * blockDim.x + threadIdx.x; idx < total; idx += gridDim.x * blockDim.x) {
    int n = idx / KPAD, k = idx % KPAD;
    float v = 0.f;
    if (k < K) v = transpose ? src[(size_t)k * N + n] : src[(size_t)n * K + k];
    dst[idx] = (_Float16)(v * scale);
  }
}

// ================= K1: per-node irrep layernorm + onehot projections ==========
__global__ __launch_bounds__(128) void node_prep(
    const float* __restrict__ nf, const float* __restrict__ oh,
    const float* __restrict__ g0, const float* __restrict__ b0,
    const float* __restrict__ g1, const float* __restrict__ g2,
    const float* __restrict__ ohw0, const float* __restrict__ ohw1, const float* __restrict__ ohw2,
    float* __restrict__ nnorm, float* __restrict__ ohp) {
  int lane = threadIdx.x & 31;
  int wid  = threadIdx.x >> 5;
  int n = blockIdx.x * 4 + wid;
  if (n >= NODES) return;
  const float* x = nf + (size_t)n * NDIM;
  float s = 0.f, ss = 0.f;
  for (int j = lane; j < 64; j += 32) { float v = x[j]; s += v; ss += v * v; }
  for (int o = 16; o; o >>= 1) { s += __shfl_xor(s, o, 32); ss += __shfl_xor(ss, o, 32); }
  float mu = s * (1.f / 64.f);
  float inv0 = rsqrtf(ss * (1.f / 64.f) - mu * mu + 1e-8f);
  float q1 = 0.f, q2 = 0.f;
  for (int j = lane; j < 96; j += 32) { float v = x[64 + j];  q1 += v * v; }
  for (int j = lane; j < 80; j += 32) { float v = x[160 + j]; q2 += v * v; }
  for (int o = 16; o; o >>= 1) { q1 += __shfl_xor(q1, o, 32); q2 += __shfl_xor(q2, o, 32); }
  float inv12 = rsqrtf(0.5f * (q1 * (1.f / 96.f) + q2 * (1.f / 80.f)) + 1e-8f);
  float* y = nnorm + (size_t)n * NDIM;
  for (int j = lane; j < 64; j += 32) y[j]       = (x[j] - mu) * inv0 * g0[j] + b0[j];
  for (int j = lane; j < 96; j += 32) y[64 + j]  = x[64 + j]  * inv12 * g1[j / 3];
  for (int j = lane; j < 80; j += 32) y[160 + j] = x[160 + j] * inv12 * g2[j / 5];
  __shared__ float soh[4][OHD];
  for (int j = lane; j < OHD; j += 32) soh[wid][j] = oh[(size_t)n * OHD + j];
  asm volatile("s_wait_dscnt 0" ::: "memory");
  const float sc = 0.08838834764831845f;  // 1/sqrt(128)
  for (int o = lane; o < 112; o += 32) {
    const float* w = (o < 64) ? (ohw0 + (size_t)o * OHD)
                              : (o < 96 ? ohw1 + (size_t)(o - 64) * OHD
                                        : ohw2 + (size_t)(o - 96) * OHD);
    float acc = 0.f;
    for (int j = 0; j < OHD; ++j) acc += soh[wid][j] * w[j];
    ohp[(size_t)n * 112 + o] = acc * sc;
  }
}

// ================= K2: per-edge mega-kernel (1 wave = 16 edges) ===============
struct EdgeP {
  const float* latents;  const float* edge_features;
  const int* edge_index; const int* active_edges;
  const float* edge_one_hot; const float* wigner; const float* cutoff;
  const float* e_g0; const float* e_b0; const float* e_g1; const float* e_g2;
  const float* post_b0; const float* ln_g; const float* ln_b;
  const float* nnorm; float* acc; const _Float16* wpool; float* out_lat;
};

__global__ __launch_bounds__(32) void edge_kernel(EdgeP P) {
  int lane = threadIdx.x;
  int m  = lane & 15;          // edge within tile
  int pr = lane >> 4;          // part 0/1 for elementwise work
  int e  = blockIdx.x * 16 + m;
  int ae = P.active_edges[e];
  int src = P.edge_index[ae];
  int dst = P.edge_index[EDGES + ae];

  __shared__ _Float16 SAbig[16][480];  // m0_in: x0|x1m1|x2m2|lat ; later H stages
  __shared__ _Float16 SApos[16][160];  // pos1 ; later Y1 stage / silu stages
  __shared__ _Float16 SAneg[16][160];  // neg1 ; later Y2 stage / silu stages
  __shared__ _Float16 SA2p[16][64];    // x2 m=+2 ; later silu(y0) stage
  __shared__ _Float16 SA2n[16][64];    // x2 m=-2
  __shared__ float SM0[16][160];       // m0 ; later z0(0..63)+z1(64..159)
  __shared__ float SR2[16][96];        // o_pos(0..47)|o_neg(48..95) ; later z2 (5x16)
  __shared__ float SQ[16][32];         // q_pos(0..15)|q_neg(16..31)
  __shared__ float SH[16][128];        // MLP results ; finally env weights (112)
  __shared__ float SD[16][36];         // D1 (9) + D2 (25)

  // ---- Phase A: Wigner blocks, layernorms, gathers, rotations, A staging ----
  const float* W = P.wigner + (size_t)e * 81;
  for (int i = pr; i < 9;  i += 2) SD[m][i]     = W[(1 + i / 3) * 9 + 1 + i % 3];
  for (int i = pr; i < 25; i += 2) SD[m][9 + i] = W[(4 + i / 5) * 9 + 4 + i % 5];

  const float* ef = P.edge_features + (size_t)e * NDIM;
  float s = 0.f, ss = 0.f;
  for (int j = pr * 32; j < pr * 32 + 32; ++j) { float v = ef[j]; s += v; ss += v * v; }
  s += __shfl_xor(s, 16, 32); ss += __shfl_xor(ss, 16, 32);
  float mu = s * (1.f / 64.f);
  float inv0 = rsqrtf(ss * (1.f / 64.f) - mu * mu + 1e-8f);
  float q1 = 0.f, q2 = 0.f;
  for (int j = pr; j < 96; j += 2) { float v = ef[64 + j];  q1 += v * v; }
  for (int j = pr; j < 80; j += 2) { float v = ef[160 + j]; q2 += v * v; }
  q1 += __shfl_xor(q1, 16, 32); q2 += __shfl_xor(q2, 16, 32);
  float inve = rsqrtf(0.5f * (q1 * (1.f / 96.f) + q2 * (1.f / 80.f)) + 1e-8f);

  const float* ns = P.nnorm + (size_t)src * NDIM;
  const float* nd = P.nnorm + (size_t)dst * NDIM;
  for (int j = pr; j < 64; j += 2) {
    SAbig[m][j]       = (_Float16)ns[j];
    SAbig[m][64 + j]  = (_Float16)((ef[j] - mu) * inv0 * P.e_g0[j] + P.e_b0[j]);
    SAbig[m][128 + j] = (_Float16)nd[j];
  }
  const float* lat = P.latents + (size_t)ae * LATD;
  for (int j = pr; j < LATD; j += 2) SAbig[m][336 + j] = (_Float16)lat[j];
  for (int j = 464 + pr; j < 480; j += 2) SAbig[m][j] = (_Float16)0.f;
  for (int j = 144 + pr; j < 160; j += 2) { SApos[m][j] = (_Float16)0.f; SAneg[m][j] = (_Float16)0.f; }
  for (int j = 48 + pr;  j < 64;  j += 2) { SA2p[m][j] = (_Float16)0.f;  SA2n[m][j] = (_Float16)0.f; }
  asm volatile("s_wait_dscnt 0" ::: "memory");

  for (int c = pr; c < 96; c += 2) {       // l=1 rotation (96 channels)
    float v0, v1, v2;
    if (c < 32)      { v0 = ns[64 + c * 3]; v1 = ns[64 + c * 3 + 1]; v2 = ns[64 + c * 3 + 2]; }
    else if (c < 64) { int cc = c - 32; float g = P.e_g1[cc] * inve;
                       v0 = ef[64 + cc * 3] * g; v1 = ef[64 + cc * 3 + 1] * g; v2 = ef[64 + cc * 3 + 2] * g; }
    else             { int cc = c - 64; v0 = nd[64 + cc * 3]; v1 = nd[64 + cc * 3 + 1]; v2 = nd[64 + cc * 3 + 2]; }
    SAneg[m][c]       = (_Float16)(SD[m][0] * v0 + SD[m][1] * v1 + SD[m][2] * v2);
    SAbig[m][192 + c] = (_Float16)(SD[m][3] * v0 + SD[m][4] * v1 + SD[m][5] * v2);
    SApos[m][c]       = (_Float16)(SD[m][6] * v0 + SD[m][7] * v1 + SD[m][8] * v2);
  }
  for (int c = pr; c < 48; c += 2) {       // l=2 rotation (48 channels)
    float v[5];
    if (c < 16)      { for (int j = 0; j < 5; ++j) v[j] = ns[160 + c * 5 + j]; }
    else if (c < 32) { int cc = c - 16; float g = P.e_g2[cc] * inve;
                       for (int j = 0; j < 5; ++j) v[j] = ef[160 + cc * 5 + j] * g; }
    else             { int cc = c - 32; for (int j = 0; j < 5; ++j) v[j] = nd[160 + cc * 5 + j]; }
    float r[5];
#pragma unroll
    for (int i = 0; i < 5; ++i) { float a = 0.f; for (int j = 0; j < 5; ++j) a += SD[m][9 + i * 5 + j] * v[j]; r[i] = a; }
    SA2n[m][c]        = (_Float16)r[0];
    SAneg[m][96 + c]  = (_Float16)r[1];
    SAbig[m][288 + c] = (_Float16)r[2];
    SApos[m][96 + c]  = (_Float16)r[3];
    SA2p[m][c]        = (_Float16)r[4];
  }
  asm volatile("s_wait_dscnt 0" ::: "memory");

  // ---- Phase B: SO(2) block GEMMs ----
  const _Float16* wp = P.wpool;
  gemm (&SAbig[0][0], 480, wp + OFF_W0, 480, &SM0[0][0], 160, 160, nullptr, lane);
  gemm2(&SApos[0][0], &SAneg[0][0], 160, wp + OFF_W1R, wp + OFF_W1IN, 160, &SR2[0][0],  96, 48, lane); // o_pos
  gemm2(&SApos[0][0], &SAneg[0][0], 160, wp + OFF_W1I, wp + OFF_W1R,  160, &SR2[0][48], 96, 48, lane); // o_neg
  gemm2(&SA2p[0][0],  &SA2n[0][0],  64,  wp + OFF_W2R, wp + OFF_W2IN, 64,  &SQ[0][0],   32, 16, lane); // q_pos
  gemm2(&SA2p[0][0],  &SA2n[0][0],  64,  wp + OFF_W2I, wp + OFF_W2R,  64,  &SQ[0][16],  32, 16, lane); // q_neg
  asm volatile("s_wait_dscnt 0" ::: "memory");

  // ---- Phase C: back-rotate, gate, stage for post GEMMs ----
  for (int j = pr; j < 64; j += 2) SA2p[m][j] = (_Float16)siluf(SM0[m][j]);   // s = silu(y0[:64])
  for (int c = pr; c < 32; c += 2) {                                           // y1 (D1^T) * gate
    float y0_ = SR2[m][48 + c], y1_ = SM0[m][112 + c], y2_ = SR2[m][c];
    float g = sigf(SM0[m][64 + c]);
    SApos[m][c]      = (_Float16)((SD[m][0] * y0_ + SD[m][3] * y1_ + SD[m][6] * y2_) * g);
    SApos[m][32 + c] = (_Float16)((SD[m][1] * y0_ + SD[m][4] * y1_ + SD[m][7] * y2_) * g);
    SApos[m][64 + c] = (_Float16)((SD[m][2] * y0_ + SD[m][5] * y1_ + SD[m][8] * y2_) * g);
  }
  for (int j = pr; j < 80; j += 2) { int mm = j / 16, c = j % 16; SAneg[m][mm * 32 + 16 + c] = (_Float16)0.f; }
  for (int c = pr; c < 16; c += 2) {                                           // y2 (D2^T) * gate
    float yv[5] = { SQ[m][16 + c], SR2[m][80 + c], SM0[m][144 + c], SR2[m][32 + c], SQ[m][c] };
    float g = sigf(SM0[m][96 + c]);
#pragma unroll
    for (int i = 0; i < 5; ++i) {
      float a = 0.f;
      for (int j = 0; j < 5; ++j) a += SD[m][9 + j * 5 + i] * yv[j];
      SAneg[m][i * 32 + c] = (_Float16)(a * g);
    }
  }
  asm volatile("s_wait_dscnt 0" ::: "memory");

  gemm(&SA2p[0][0], 64, wp + OFF_PW0, 64, &SM0[0][0], 160, 64, P.post_b0, lane);       // z0
  for (int mm = 0; mm < 3; ++mm)                                                        // z1
    gemm(&SApos[0][mm * 32], 160, wp + OFF_PW1, 32, &SM0[0][64 + mm * 32], 160, 32, nullptr, lane);
  for (int mm = 0; mm < 5; ++mm)                                                        // z2
    gemm(&SAneg[0][mm * 32], 160, wp + OFF_PW2, 32, &SR2[0][mm * 16], 96, 16, nullptr, lane);
  asm volatile("s_wait_dscnt 0" ::: "memory");

  // ---- Phase D: latent MLP stack ----
  float ls = 0.f, lss = 0.f;
  for (int j = pr; j < LATD; j += 2) { float v = lat[j]; ls += v; lss += v * v; }
  ls += __shfl_xor(ls, 16, 32); lss += __shfl_xor(lss, 16, 32);
  float lmu = ls * (1.f / 128.f);
  float linv = rsqrtf(lss * (1.f / 128.f) - lmu * lmu + 1e-5f);
  for (int j = pr; j < LATD; j += 2) SAbig[m][j] = (_Float16)((lat[j] - lmu) * linv * P.ln_g[j] + P.ln_b[j]);
  for (int j = pr; j < 64; j += 2)  SAbig[m][128 + j] = (_Float16)SM0[m][j];  // concat z0
  asm volatile("s_wait_dscnt 0" ::: "memory");
  gemm(&SAbig[0][0], 480, wp + OFF_M1W1, 192, &SH[0][0], 128, 128, nullptr, lane);
  asm volatile("s_wait_dscnt 0" ::: "memory");
  for (int j = pr; j < 128; j += 2) SApos[m][j] = (_Float16)siluf(SH[m][j]);
  asm volatile("s_wait_dscnt 0" ::: "memory");
  gemm(&SApos[0][0], 160, wp + OFF_M1W2, 128, &SH[0][0], 128, 128, nullptr, lane);
  asm volatile("s_wait_dscnt 0" ::: "memory");
  for (int j = pr; j < 128; j += 2) SAneg[m][j] = (_Float16)siluf(SH[m][j]);
  asm volatile("s_wait_dscnt 0" ::: "memory");
  gemm(&SAneg[0][0], 160, wp + OFF_M1W3, 128, &SH[0][0], 128, 128, nullptr, lane);
  asm volatile("s_wait_dscnt 0" ::: "memory");
  const float* eoh = P.edge_one_hot + (size_t)e * OHD;
  for (int j = pr; j < 128; j += 2) {
    SAbig[m][j]       = (_Float16)SH[m][j];
    SAbig[m][128 + j] = (_Float16)eoh[j];
  }
  asm volatile("s_wait_dscnt 0" ::: "memory");
  gemm(&SAbig[0][0], 480, wp + OFF_M2W1, 256, &SH[0][0], 128, 128, nullptr, lane);
  asm volatile("s_wait_dscnt 0" ::: "memory");
  for (int j = pr; j < 128; j += 2) SApos[m][j] = (_Float16)siluf(SH[m][j]);
  asm volatile("s_wait_dscnt 0" ::: "memory");
  gemm(&SApos[0][0], 160, wp + OFF_M2W2, 128, &SH[0][0], 128, 128, nullptr, lane);
  asm volatile("s_wait_dscnt 0" ::: "memory");
  for (int j = pr; j < 128; j += 2) SAneg[m][j] = (_Float16)siluf(SH[m][j]);
  asm volatile("s_wait_dscnt 0" ::: "memory");
  gemm(&SAneg[0][0], 160, wp + OFF_M2W3, 128, &SH[0][0], 128, 128, nullptr, lane);
  asm volatile("s_wait_dscnt 0" ::: "memory");
  float co = P.cutoff[ae];
  float* outl = P.out_lat + (size_t)e * LATD;
  for (int j = pr; j < 128; j += 2) {
    float v = SH[m][j] * co;         // new_latents
    outl[j] = v;
    SApos[m][j] = (_Float16)v;
  }
  asm volatile("s_wait_dscnt 0" ::: "memory");
  gemm(&SApos[0][0], 160, wp + OFF_ENV, 128, &SH[0][0], 128, 112, nullptr, lane);  // env weights
  asm volatile("s_wait_dscnt 0" ::: "memory");

  // ---- Phase E: gate by env weights, scatter-add into node accumulator ----
  const float nrm = 0.31622776601683794f;  // 1/sqrt(10)
  float* accrow = P.acc + (size_t)src * NDIM;
  for (int j = pr; j < 64; j += 2)
    atomicAdd(&accrow[j], SM0[m][j] * SH[m][j] * nrm);
  for (int j = pr; j < 96; j += 2) {
    int c = j / 3, mm = j % 3;
    atomicAdd(&accrow[64 + j], SM0[m][64 + mm * 32 + c] * SH[m][64 + c] * nrm);
  }
  for (int j = pr; j < 80; j += 2) {
    int c = j / 5, mm = j % 5;
    atomicAdd(&accrow[160 + j], SR2[m][mm * 16 + c] * SH[m][96 + c] * nrm);
  }
}

// ================= K3: final node blend =======================================
__global__ void node_final(const float* __restrict__ nf, const float* __restrict__ acc,
                           const float* __restrict__ ohp, float* __restrict__ out) {
  int i = blockIdx.x * blockDim.x + threadIdx.x;
  if (i >= NODES * NDIM) return;
  int n = i / NDIM, j = i % NDIM;
  int o = (j < 64) ? j : (j < 160 ? 64 + (j - 64) / 3 : 96 + (j - 160) / 5);
  float a = acc[i] * ohp[(size_t)n * 112 + o];
  out[i] = 0.5f * nf[i] + 0.5f * a;   // r = sigmoid(0) = 0.5
}

// ================= host launcher ==============================================
extern "C" void kernel_launch(void* const* d_in, const int* in_sizes, int n_in,
                              void* d_out, int out_size, void* d_ws, size_t ws_size,
                              hipStream_t stream) {
  const float* latents       = (const float*)d_in[0];
  const float* node_features = (const float*)d_in[1];
  const float* edge_features = (const float*)d_in[2];
  const float* node_onehot   = (const float*)d_in[4];
  const int*   edge_index    = (const int*)d_in[5];
  const int*   active_edges  = (const int*)d_in[7];
  const float* edge_one_hot  = (const float*)d_in[8];
  const float* wigner        = (const float*)d_in[9];
  const float* cutoff        = (const float*)d_in[10];
  // params (dict order)
  const float* sln_n_g0 = (const float*)d_in[11];
  const float* sln_n_b0 = (const float*)d_in[12];
  const float* sln_n_g1 = (const float*)d_in[13];
  const float* sln_n_g2 = (const float*)d_in[14];
  const float* sln_e_g0 = (const float*)d_in[15];
  const float* sln_e_b0 = (const float*)d_in[16];
  const float* sln_e_g1 = (const float*)d_in[17];
  const float* sln_e_g2 = (const float*)d_in[18];
  const float* post_b0  = (const float*)d_in[25];
  const float* ln_g     = (const float*)d_in[28];
  const float* ln_b     = (const float*)d_in[29];
  const float* oh_w0    = (const float*)d_in[37];
  const float* oh_w1    = (const float*)d_in[38];
  const float* oh_w2    = (const float*)d_in[39];

  float* nnorm = (float*)d_ws;
  float* acc   = nnorm + (size_t)NODES * NDIM;
  float* ohp   = acc   + (size_t)NODES * NDIM;
  _Float16* wpool = (_Float16*)(ohp + (size_t)NODES * 112);
  float* out_node = (float*)d_out;
  float* out_lat  = out_node + (size_t)NODES * NDIM;

  auto prep = [&](int src_idx, int off, int N, int K, int KPAD, int tr, float sc) {
    int total = N * KPAD;
    int blocks = (total + 255) / 256;
    prep_w<<<blocks, 256, 0, stream>>>((const float*)d_in[src_idx], wpool + off, N, K, KPAD, tr, sc);
  };
  prep(19, OFF_W0,   160, 464, 480, 0,  1.f);  // so2_w0
  prep(20, OFF_W1R,  48,  144, 160, 0,  1.f);
  prep(21, OFF_W1I,  48,  144, 160, 0,  1.f);
  prep(21, OFF_W1IN, 48,  144, 160, 0, -1.f);
  prep(22, OFF_W2R,  16,  48,  64,  0,  1.f);
  prep(23, OFF_W2I,  16,  48,  64,  0,  1.f);
  prep(23, OFF_W2IN, 16,  48,  64,  0, -1.f);
  prep(24, OFF_PW0,  64,  64,  64,  0,  1.f);  // post_w0
  prep(26, OFF_PW1,  32,  32,  32,  0,  1.f);  // post_w1
  prep(27, OFF_PW2,  16,  16,  32,  0,  1.f);  // post_w2
  prep(30, OFF_M1W1, 128, 192, 192, 1,  1.f);  // mlp1_w1 (transpose in-major)
  prep(31, OFF_M1W2, 128, 128, 128, 1,  1.f);
  prep(32, OFF_M1W3, 128, 128, 128, 1,  1.f);
  prep(33, OFF_M2W1, 128, 256, 256, 1,  1.f);
  prep(34, OFF_M2W2, 128, 128, 128, 1,  1.f);
  prep(35, OFF_M2W3, 128, 128, 128, 1,  1.f);
  prep(36, OFF_ENV,  112, 128, 128, 1,  1.f);  // env_w

  hipMemsetAsync(acc, 0, (size_t)NODES * NDIM * sizeof(float), stream);

  node_prep<<<NODES / 4, 128, 0, stream>>>(node_features, node_onehot,
      sln_n_g0, sln_n_b0, sln_n_g1, sln_n_g2, oh_w0, oh_w1, oh_w2, nnorm, ohp);

  EdgeP ep;
  ep.latents = latents; ep.edge_features = edge_features;
  ep.edge_index = edge_index; ep.active_edges = active_edges;
  ep.edge_one_hot = edge_one_hot; ep.wigner = wigner; ep.cutoff = cutoff;
  ep.e_g0 = sln_e_g0; ep.e_b0 = sln_e_b0; ep.e_g1 = sln_e_g1; ep.e_g2 = sln_e_g2;
  ep.post_b0 = post_b0; ep.ln_g = ln_g; ep.ln_b = ln_b;
  ep.nnorm = nnorm; ep.acc = acc; ep.wpool = wpool; ep.out_lat = out_lat;
  edge_kernel<<<EDGES / 16, 32, 0, stream>>>(ep);   // 6250 exact 16-edge tiles

  node_final<<<(NODES * NDIM + 255) / 256, 256, 0, stream>>>(node_features, acc, ohp, out_node);
}
